// Attention_34454227649319
// MI455X (gfx1250) — compile-verified, matrix operand order checked by reading, not run on hardware
//
#include <hip/hip_runtime.h>
#include <hip/hip_bf16.h>

// ---------------- problem constants ----------------
constexpr int Bc   = 4;
constexpr int Sc   = 2048;
constexpr int HIDc = 2048;
constexpr int NHc  = 16;
constexpr int NKVc = 4;
constexpr int HDc  = 128;

typedef __attribute__((ext_vector_type(16))) _Float16 v16h;
typedef __attribute__((ext_vector_type(8)))  _Float16 v8h;
typedef __attribute__((ext_vector_type(8)))  float    v8f;
typedef __attribute__((ext_vector_type(4)))  int      v4i;

// ---- gfx1250 async global->LDS copy path (guarded; falls back cleanly) ----
#if defined(__has_builtin)
#  if __has_builtin(__builtin_amdgcn_global_load_async_to_lds_b128) && \
      __has_builtin(__builtin_amdgcn_s_wait_asynccnt)
#    define USE_ASYNC_LDS 1
#  endif
#endif
#ifndef USE_ASYNC_LDS
#  define USE_ASYNC_LDS 0
#endif

#if USE_ASYNC_LDS
__device__ __forceinline__ void async_cp16(const _Float16* gsrc, _Float16* ldst) {
  // Builtin expects (global v4i*, local v4i*, imm offset, imm cpol).
  // generic->AS1 is identity; generic LDS ptr low 32 bits are the LDS offset.
  __builtin_amdgcn_global_load_async_to_lds_b128(
      (__attribute__((address_space(1))) v4i*)(uintptr_t)gsrc,
      (__attribute__((address_space(3))) v4i*)(unsigned int)(uintptr_t)ldst,
      0, 0);
}
#endif

// Load one 16x32 f16 A/B fragment row (16 halfs) from LDS per CDNA5 layout:
// element i <-> k = kb + (i<8 ? i : i+8); kb = (lane<16 ? 0 : 8).
__device__ __forceinline__ v16h ld_frag(const _Float16* p, int kb) {
  v8h lo = *(const v8h*)(p + kb);
  v8h hi = *(const v8h*)(p + kb + 16);
  return __builtin_shufflevector(lo, hi, 0,1,2,3,4,5,6,7,8,9,10,11,12,13,14,15);
}

// ---------------- elementwise converts ----------------
__global__ void f32_to_f16_kernel(const float* __restrict__ in,
                                  _Float16* __restrict__ out, size_t n) {
  size_t i = (size_t)blockIdx.x * blockDim.x + threadIdx.x;
  if (i < n) out[i] = (_Float16)in[i];
}

// in: W[K][N] row-major f32 -> out: WT[N][K] row-major f16 (tiled, coalesced)
__global__ __launch_bounds__(256) void transpose_cvt_kernel(
    const float* __restrict__ in, _Float16* __restrict__ out, int K, int N) {
  __shared__ _Float16 tile[32][33];
  const int t  = threadIdx.x;
  const int bk = blockIdx.y * 32;   // K tile
  const int bn = blockIdx.x * 32;   // N tile
  for (int i = 0; i < 4; ++i) {
    int idx = t + i * 256;
    int r = idx >> 5, c = idx & 31;
    tile[r][c] = (_Float16)in[(size_t)(bk + r) * N + bn + c];
  }
  __syncthreads();
  for (int i = 0; i < 4; ++i) {
    int idx = t + i * 256;
    int r = idx >> 5, c = idx & 31;
    out[(size_t)(bn + r) * K + bk + c] = tile[c][r];
  }
}

// ---------------- RoPE (interleaved even/odd pairs) ----------------
__global__ void rope_kernel(_Float16* __restrict__ p, const float* __restrict__ cosb,
                            const float* __restrict__ sinb, int nheads) {
  size_t idx = (size_t)blockIdx.x * blockDim.x + threadIdx.x;
  size_t total = (size_t)Bc * Sc * nheads * (HDc / 2);
  if (idx >= total) return;
  int d2 = (int)(idx % (HDc / 2));
  int hh = (int)((idx / (HDc / 2)) % nheads);
  int s  = (int)((idx / ((size_t)(HDc / 2) * nheads)) % Sc);
  int b  = (int)(idx / ((size_t)(HDc / 2) * nheads * Sc));
  size_t base = (((size_t)b * Sc + s) * nheads + hh) * HDc + 2 * d2;
  float u0 = (float)p[base];
  float u1 = (float)p[base + 1];
  float c  = cosb[(size_t)s * HDc + 2 * d2];
  float sn = sinb[(size_t)s * HDc + 2 * d2];
  p[base]     = (_Float16)(u0 * c - u1 * sn);
  p[base + 1] = (_Float16)(u1 * c + u0 * sn);
}

// ---------------- WMMA GEMM: C[M,N] = A[M,K] * BT[N,K]^T ----------------
// 128x128 block tile, 256 threads = 8 waves (4 m-waves x 2 n-waves),
// wave tile 32x64 = 2x4 fragments. Double-buffered LDS, one barrier per
// 32-wide K step; next tile staged with async global->LDS copies when
// available, else a load-reg/compute/store-LDS software pipeline.
template<bool OUTF32>
__global__ __launch_bounds__(256) void gemm_wmma_kernel(
    const _Float16* __restrict__ A, const _Float16* __restrict__ BT,
    void* __restrict__ Cout, int M, int N, int K) {
  __shared__ _Float16 sA[2][128 * 40];   // 32 cols + 8 pad (rows 80B apart)
  __shared__ _Float16 sB[2][128 * 40];

  const int t    = threadIdx.x;
  const int lane = t & 31;
  const int wid  = t >> 5;
  const int wm   = wid & 3;      // 0..3 -> 32 rows each
  const int wn   = wid >> 2;     // 0..1 -> 64 cols each
  const int lr   = lane & 15;
  const int lh   = lane >> 4;
  const int kb   = (lane < 16) ? 0 : 8;
  const int bm   = blockIdx.y * 128;
  const int bn   = blockIdx.x * 128;

  v8f acc[2][4] = {};

  // ---- prologue: stage K-step 0 into buffer 0 ----
#if USE_ASYNC_LDS
  for (int i = 0; i < 2; ++i) {
    int ch = i * 256 + t, row = ch >> 2, c = (ch & 3) * 8;
    async_cp16(&A[(size_t)(bm + row) * K + c],  &sA[0][row * 40 + c]);
    async_cp16(&BT[(size_t)(bn + row) * K + c], &sB[0][row * 40 + c]);
  }
  __builtin_amdgcn_s_wait_asynccnt(0);
  __syncthreads();
#else
  for (int i = 0; i < 2; ++i) {
    int ch = i * 256 + t, row = ch >> 2, c = (ch & 3) * 8;
    *(v8h*)&sA[0][row * 40 + c] = *(const v8h*)&A[(size_t)(bm + row) * K + c];
    *(v8h*)&sB[0][row * 40 + c] = *(const v8h*)&BT[(size_t)(bn + row) * K + c];
  }
  __syncthreads();
#endif

  int buf = 0;
  for (int k0 = 0; k0 < K; k0 += 32) {
    const int  kn       = k0 + 32;
    const bool has_next = kn < K;

#if USE_ASYNC_LDS
    if (has_next)
      for (int i = 0; i < 2; ++i) {
        int ch = i * 256 + t, row = ch >> 2, c = (ch & 3) * 8;
        async_cp16(&A[(size_t)(bm + row) * K + kn + c],  &sA[buf ^ 1][row * 40 + c]);
        async_cp16(&BT[(size_t)(bn + row) * K + kn + c], &sB[buf ^ 1][row * 40 + c]);
      }
#else
    v8h ra[2], rb[2];
    if (has_next)
      for (int i = 0; i < 2; ++i) {
        int ch = i * 256 + t, row = ch >> 2, c = (ch & 3) * 8;
        ra[i] = *(const v8h*)&A[(size_t)(bm + row) * K + kn + c];
        rb[i] = *(const v8h*)&BT[(size_t)(bn + row) * K + kn + c];
      }
#endif

    // ---- compute from current buffer (overlaps with staging above) ----
    v16h af[2];
#pragma unroll
    for (int fm = 0; fm < 2; ++fm)
      af[fm] = ld_frag(&sA[buf][(wm * 32 + fm * 16 + lr) * 40], kb);
#pragma unroll
    for (int fn = 0; fn < 4; ++fn) {
      v16h bf = ld_frag(&sB[buf][(wn * 64 + fn * 16 + lr) * 40], kb);
#pragma unroll
      for (int fm = 0; fm < 2; ++fm)
        acc[fm][fn] = __builtin_amdgcn_wmma_f32_16x16x32_f16(
            false, af[fm], false, bf, (short)0, acc[fm][fn], false, false);
    }

#if USE_ASYNC_LDS
    __builtin_amdgcn_s_wait_asynccnt(0);
#else
    if (has_next)
      for (int i = 0; i < 2; ++i) {
        int ch = i * 256 + t, row = ch >> 2, c = (ch & 3) * 8;
        *(v8h*)&sA[buf ^ 1][row * 40 + c] = ra[i];
        *(v8h*)&sB[buf ^ 1][row * 40 + c] = rb[i];
      }
#endif
    __syncthreads();
    buf ^= 1;
  }

  // ---- epilogue (branch-free: output type is a template parameter) ----
#pragma unroll
  for (int fm = 0; fm < 2; ++fm)
#pragma unroll
    for (int fn = 0; fn < 4; ++fn) {
      const int n = bn + wn * 64 + fn * 16 + lr;
#pragma unroll
      for (int r = 0; r < 8; ++r) {
        const int m = bm + wm * 32 + fm * 16 + r + 8 * lh;
        const float v = acc[fm][fn][r];
        if constexpr (OUTF32)
          ((float*)Cout)[(size_t)m * N + n] = v;
        else
          ((_Float16*)Cout)[(size_t)m * N + n] = (_Float16)v;
      }
    }
}

// ---------------- flash attention (causal, GQA) ----------------
// One block per (b, head, 64-row q tile). 128 threads = 4 waves, each wave
// owns 16 q rows. QK^T and PV both via WMMA f16; online softmax in fp32.
__global__ __launch_bounds__(128) void attn_kernel(
    const _Float16* __restrict__ Q, const _Float16* __restrict__ Kv,
    const _Float16* __restrict__ Vv, _Float16* __restrict__ O) {
  __shared__ _Float16 sQ[64 * 136];   // [q][hd], 128+8 pad
  __shared__ _Float16 sK[64 * 136];   // [key][hd]
  __shared__ _Float16 sV[128 * 72];   // transposed: [hd][key], 64+8 pad
  __shared__ _Float16 sP[64 * 72];    // [q][key]

  const int t    = threadIdx.x;
  const int lane = t & 31;
  const int w    = t >> 5;            // wave 0..3 -> q rows w*16..w*16+15
  const int lr   = lane & 15;
  const int lh   = lane >> 4;
  const int kb   = (lane < 16) ? 0 : 8;

  const int nqt = Sc / 64;
  int blk = blockIdx.x;
  int qt  = blk % nqt;
  int h   = (blk / nqt) % NHc;
  int b   = blk / (nqt * NHc);
  int g   = h / (NHc / NKVc);
  int qbase = qt * 64;

  // stage Q tile
  for (int idx = t; idx < 64 * 16; idx += 128) {
    int row = idx >> 4;
    int c   = (idx & 15) * 8;
    int q   = qbase + row;
    *(v8h*)&sQ[row * 136 + c] =
        *(const v8h*)&Q[(((size_t)b * Sc + q) * NHc + h) * HDc + c];
  }
  __syncthreads();

  v16h aq[4];
#pragma unroll
  for (int kk = 0; kk < 4; ++kk)
    aq[kk] = ld_frag(&sQ[(w * 16 + lr) * 136 + kk * 32], kb);

  v8f  oacc[8] = {};
  float rm[8], rl[8];
#pragma unroll
  for (int r = 0; r < 8; ++r) { rm[r] = -3.0e38f; rl[r] = 0.0f; }

  const float scale = 0.08838834764831845f;  // 1/sqrt(128)

  for (int j = 0; j <= qt; ++j) {
    __syncthreads();
    // stage K tile [64 keys][128 hd] and V tile transposed [128 hd][64 keys]
    for (int idx = t; idx < 64 * 16; idx += 128) {
      int row = idx >> 4;
      int c   = (idx & 15) * 8;
      int key = j * 64 + row;
      size_t gbase = (((size_t)b * Sc + key) * NKVc + g) * HDc + c;
      *(v8h*)&sK[row * 136 + c] = *(const v8h*)&Kv[gbase];
      v8h vv = *(const v8h*)&Vv[gbase];
#pragma unroll
      for (int e = 0; e < 8; ++e) sV[(c + e) * 72 + row] = vv[e];
      if (j < qt) {  // prefetch next tile into cache (global_prefetch_b8)
        size_t gnext = gbase + (size_t)64 * NKVc * HDc;
        __builtin_prefetch(&Kv[gnext], 0, 1);
        __builtin_prefetch(&Vv[gnext], 0, 1);
      }
    }
    __syncthreads();

    // scores: 16 q rows x 64 keys per wave
    v8f sacc[4] = {};
#pragma unroll
    for (int nt = 0; nt < 4; ++nt)
#pragma unroll
      for (int kk = 0; kk < 4; ++kk) {
        v16h bk = ld_frag(&sK[(nt * 16 + lr) * 136 + kk * 32], kb);
        sacc[nt] = __builtin_amdgcn_wmma_f32_16x16x32_f16(
            false, aq[kk], false, bk, (short)0, sacc[nt], false, false);
      }

    const bool diag = (j == qt);
#pragma unroll
    for (int r = 0; r < 8; ++r) {
      int qrow = qbase + w * 16 + r + 8 * lh;
      float mx = -3.0e38f;
#pragma unroll
      for (int nt = 0; nt < 4; ++nt) {
        float s = sacc[nt][r] * scale;
        if (diag) {
          int key = j * 64 + nt * 16 + lr;
          if (key > qrow) s = -3.0e38f;
        }
        sacc[nt][r] = s;
        mx = fmaxf(mx, s);
      }
      // reduce across the 16 lanes of this half (wave32, xor masks < 16)
      for (int off = 8; off; off >>= 1) mx = fmaxf(mx, __shfl_xor(mx, off, 32));
      float mnew  = fmaxf(rm[r], mx);
      float alpha = __expf(rm[r] - mnew);
      rm[r] = mnew;
      float rs = 0.0f;
#pragma unroll
      for (int nt = 0; nt < 4; ++nt) {
        float pv = __expf(sacc[nt][r] - mnew);
        rs += pv;
        sP[(w * 16 + r + 8 * lh) * 72 + nt * 16 + lr] = (_Float16)pv;
      }
      for (int off = 8; off; off >>= 1) rs += __shfl_xor(rs, off, 32);
      rl[r] = rl[r] * alpha + rs;
#pragma unroll
      for (int ht = 0; ht < 8; ++ht) oacc[ht][r] *= alpha;
    }
    __syncthreads();  // sP visible

    // O += P @ V : A = P (16 rows x 64 keys), B = V^T columns = hd
    v16h ap[2];
#pragma unroll
    for (int ks = 0; ks < 2; ++ks)
      ap[ks] = ld_frag(&sP[(w * 16 + lr) * 72 + ks * 32], kb);
#pragma unroll
    for (int ht = 0; ht < 8; ++ht)
#pragma unroll
      for (int ks = 0; ks < 2; ++ks) {
        v16h bv = ld_frag(&sV[(ht * 16 + lr) * 72 + ks * 32], kb);
        oacc[ht] = __builtin_amdgcn_wmma_f32_16x16x32_f16(
            false, ap[ks], false, bv, (short)0, oacc[ht], false, false);
      }
  }

#pragma unroll
  for (int ht = 0; ht < 8; ++ht)
#pragma unroll
    for (int r = 0; r < 8; ++r) {
      int q = qbase + w * 16 + r + 8 * lh;
      int d = ht * 16 + lr;
      O[(((size_t)b * Sc + q) * NHc + h) * HDc + d] = (_Float16)(oacc[ht][r] / rl[r]);
    }
}

// ---------------- host launcher ----------------
extern "C" void kernel_launch(void* const* d_in, const int* in_sizes, int n_in,
                              void* d_out, int out_size, void* d_ws, size_t ws_size,
                              hipStream_t stream) {
  (void)in_sizes; (void)n_in; (void)out_size; (void)ws_size;
  const float* x    = (const float*)d_in[0];
  const float* cosb = (const float*)d_in[1];
  const float* sinb = (const float*)d_in[2];
  const float* wq   = (const float*)d_in[3];
  const float* wk   = (const float*)d_in[4];
  const float* wv   = (const float*)d_in[5];
  const float* wo   = (const float*)d_in[6];
  float* out = (float*)d_out;

  const int M   = Bc * Sc;          // 8192
  const int NQ  = NHc * HDc;        // 2048
  const int NKv = NKVc * HDc;       // 512

  _Float16* p = (_Float16*)d_ws;
  _Float16* xh  = p; p += (size_t)M * HIDc;
  _Float16* wqT = p; p += (size_t)NQ * HIDc;
  _Float16* wkT = p; p += (size_t)NKv * HIDc;
  _Float16* wvT = p; p += (size_t)NKv * HIDc;
  _Float16* woT = p; p += (size_t)HIDc * NQ;
  _Float16* qh  = p; p += (size_t)M * NQ;
  _Float16* kh  = p; p += (size_t)M * NKv;
  _Float16* vh  = p; p += (size_t)M * NKv;
  _Float16* ctx = p; p += (size_t)M * NQ;

  // 1) convert x to f16
  {
    size_t n = (size_t)M * HIDc;
    f32_to_f16_kernel<<<(unsigned)((n + 255) / 256), 256, 0, stream>>>(x, xh, n);
  }
  // 2) transpose-convert weights (tiled)
  transpose_cvt_kernel<<<dim3(NQ / 32, HIDc / 32), 256, 0, stream>>>(wq, wqT, HIDc, NQ);
  transpose_cvt_kernel<<<dim3(NKv / 32, HIDc / 32), 256, 0, stream>>>(wk, wkT, HIDc, NKv);
  transpose_cvt_kernel<<<dim3(NKv / 32, HIDc / 32), 256, 0, stream>>>(wv, wvT, HIDc, NKv);
  transpose_cvt_kernel<<<dim3(HIDc / 32, NQ / 32), 256, 0, stream>>>(wo, woT, NQ, HIDc);
  // 3) QKV projections (f16 WMMA GEMMs)
  gemm_wmma_kernel<false><<<dim3(NQ / 128, M / 128), 256, 0, stream>>>(
      xh, wqT, (void*)qh, M, NQ, HIDc);
  gemm_wmma_kernel<false><<<dim3(NKv / 128, M / 128), 256, 0, stream>>>(
      xh, wkT, (void*)kh, M, NKv, HIDc);
  gemm_wmma_kernel<false><<<dim3(NKv / 128, M / 128), 256, 0, stream>>>(
      xh, wvT, (void*)vh, M, NKv, HIDc);
  // 4) RoPE on q and k
  {
    size_t nq = (size_t)Bc * Sc * NHc * (HDc / 2);
    rope_kernel<<<(unsigned)((nq + 255) / 256), 256, 0, stream>>>(qh, cosb, sinb, NHc);
    size_t nk = (size_t)Bc * Sc * NKVc * (HDc / 2);
    rope_kernel<<<(unsigned)((nk + 255) / 256), 256, 0, stream>>>(kh, cosb, sinb, NKVc);
  }
  // 5) causal flash attention
  attn_kernel<<<dim3(Bc * NHc * (Sc / 64)), 128, 0, stream>>>(qh, kh, vh, ctx);
  // 6) output projection -> fp32 d_out
  gemm_wmma_kernel<true><<<dim3(HIDc / 128, M / 128), 256, 0, stream>>>(
      ctx, woT, (void*)out, M, HIDc, HIDc);
}
